// ArithmeticModel_52029233823794
// MI455X (gfx1250) — matrix operand order repaired and checked by reading, not compiled, not run on hardware
//
#include <hip/hip_runtime.h>
#include <hip/hip_bf16.h>
#include <math.h>

// ---------------------------------------------------------------------------
// Model dims (fixed by the reference)
// ---------------------------------------------------------------------------
#define Bc   2
#define Sc   1024
#define Dc   1024
#define Hc   16
#define DFFc 4096
#define Lc   6
#define Vc   32000
#define HDc  64
#define MTOK (Bc * Sc)           // 2048 token rows

typedef __bf16 bf16_t;
typedef __attribute__((ext_vector_type(16))) __bf16 v16bf;
typedef __attribute__((ext_vector_type(8)))  __bf16 v8bf;
typedef __attribute__((ext_vector_type(4)))  __bf16 v4bf;
typedef __attribute__((ext_vector_type(8)))  float  v8f;

__device__ __forceinline__ v16bf cat8(v8bf lo, v8bf hi) {
  return __builtin_shufflevector(lo, hi, 0,1,2,3,4,5,6,7,8,9,10,11,12,13,14,15);
}

// A-fragment (16x32 bf16, M x K). ISA layout: lane (r = lane&15 -> M row,
// h = lane>>4): elements 0..7 = K(h*8 .. h*8+7), elements 8..15 = K(16+h*8 ..).
__device__ __forceinline__ v16bf load_frag_a(const bf16_t* base, int stride) {
  const int lane = threadIdx.x & 31;
  const int r = lane & 15, h = lane >> 4;
  const bf16_t* p = base + r * stride + h * 8;
  v8bf lo = *(const v8bf*)(p);
  v8bf hi = *(const v8bf*)(p + 16);
  return cat8(lo, hi);
}

// B-fragment (32x16 bf16, K x N) stored as N rows of K-contiguous data.
// ISA layout: lane (r -> N col, h): elements 0..7 = K(h*8 .. h*8+7), elements 8..15 = K(16+h*8 ..).
__device__ __forceinline__ v16bf load_frag_b(const bf16_t* base, int stride) {
  const int lane = threadIdx.x & 31;
  const int r = lane & 15, h = lane >> 4;
  const bf16_t* p = base + r * stride + h * 8;
  v8bf lo = *(const v8bf*)(p);
  v8bf hi = *(const v8bf*)(p + 16);
  return cat8(lo, hi);
}

__device__ __forceinline__ v8f wmma_bf16(v16bf a, v16bf b, v8f c) {
  return __builtin_amdgcn_wmma_f32_16x16x32_bf16(false, a, false, b,
                                                 (short)0, c, false, false);
}

// Async 16-byte global->LDS copy (CDNA5, ASYNCcnt-tracked DMA). The LDS
// destination offset is the low 32 bits of the generic shared pointer
// (flat shared addr = {SHARED_BASE,16'b0} | lds_offset).
__device__ __forceinline__ void async_ld16(const bf16_t* lds_dst, const bf16_t* g) {
  const unsigned off = (unsigned)(unsigned long long)(const void*)lds_dst;
  asm volatile("global_load_async_to_lds_b128 %0, %1, off"
               :: "v"(off), "v"(g) : "memory");
}

// ---------------------------------------------------------------------------
// GEMM: C[M,N] = A[M,K] @ W[N,K]^T (+bias) (+ReLU), A/W already bf16.
// 128x128x32 block tiles, 8 waves, wave = 64x32 of C. Double-buffered LDS
// filled by async DMA. Steady-state loop always prefetches next tile and
// issues a branch-free s_wait_asynccnt 0x4 (async loads complete in order,
// so <=4 outstanding == oldest batch of 4 landed); last tile is peeled.
// LDS rows padded to 40 bf16 (80 B = 20 banks): 16-B fragment reads from the
// 16 lanes start at banks {20r mod 64} = all multiples of 4 -> conflict-free.
// ---------------------------------------------------------------------------
#define GBM 128
#define GBN 128
#define GBK 32
#define GLS 40   // LDS row stride (bf16 elements)

// flags: bit0 = ReLU, bit1 = bf16 output, bit2 = V-transposed bf16 output
#define GF_RELU 1
#define GF_OBF  2
#define GF_VTR  4

__device__ __forceinline__ void gemm_stage(const bf16_t* __restrict__ A,
                                           const bf16_t* __restrict__ W,
                                           bf16_t* sA, bf16_t* sB,
                                           int tileM, int tileN, int K, int kb,
                                           int tid) {
  // 128x32 bf16 tile = 512 x 16-B chunks; 256 threads x 2 chunks each
  // -> 2 async instructions per wave per tile per matrix (4 total).
  #pragma unroll
  for (int i = 0; i < 2; ++i) {
    const int c    = tid + i * 256;     // 0..511
    const int row  = c >> 2;
    const int col8 = (c & 3) << 3;      // 8-element chunk start
    async_ld16(sA + row * GLS + col8, A + (size_t)(tileM + row) * K + kb + col8);
    async_ld16(sB + row * GLS + col8, W + (size_t)(tileN + row) * K + kb + col8);
  }
}

__device__ __forceinline__ void gemm_compute(const bf16_t* cA, const bf16_t* cB,
                                             int wm, int wn, v8f (&acc)[4][2]) {
  v16bf af[4], bfr[2];
  #pragma unroll
  for (int mi = 0; mi < 4; ++mi)
    af[mi] = load_frag_a(cA + (wm * 64 + mi * 16) * GLS, GLS);
  #pragma unroll
  for (int ni = 0; ni < 2; ++ni)
    bfr[ni] = load_frag_b(cB + (wn * 32 + ni * 16) * GLS, GLS);
  #pragma unroll
  for (int mi = 0; mi < 4; ++mi)
    #pragma unroll
    for (int ni = 0; ni < 2; ++ni)
      acc[mi][ni] = wmma_bf16(af[mi], bfr[ni], acc[mi][ni]);
}

template <int FLAGS>
__global__ __launch_bounds__(256)
void gemm_bf16_kernel(const bf16_t* __restrict__ A, const bf16_t* __restrict__ W,
                      const float* __restrict__ bias, void* __restrict__ Cout,
                      int M, int N, int K)
{
  __shared__ __align__(16) bf16_t sA[2 * GBM * GLS];
  __shared__ __align__(16) bf16_t sB[2 * GBN * GLS];

  const int tid   = threadIdx.x;
  const int wave  = tid >> 5;
  const int wm    = wave >> 2;              // 0..1 -> M offset wm*64
  const int wn    = wave & 3;               // 0..3 -> N offset wn*32
  const int tileM = blockIdx.y * GBM;
  const int tileN = blockIdx.x * GBN;

  v8f acc[4][2] = {};

  gemm_stage(A, W, sA, sB, tileM, tileN, K, 0, tid);   // prologue -> buf 0

  int buf = 0;
  int kb  = 0;
  for (; kb < K - GBK; kb += GBK, buf ^= 1) {          // steady state: prefetch
    gemm_stage(A, W, sA + (buf ^ 1) * GBM * GLS, sB + (buf ^ 1) * GBN * GLS,
               tileM, tileN, K, kb + GBK, tid);
    asm volatile("s_wait_asynccnt 0x4" ::: "memory");  // oldest batch landed
    __syncthreads();
    gemm_compute(sA + buf * GBM * GLS, sB + buf * GBN * GLS, wm, wn, acc);
    __syncthreads();   // readers done before next iter overwrites this buffer
  }
  // peeled final tile: no prefetch, drain everything
  asm volatile("s_wait_asynccnt 0x0" ::: "memory");
  __syncthreads();
  gemm_compute(sA + buf * GBM * GLS, sB + buf * GBN * GLS, wm, wn, acc);

  // Epilogue. C-frag layout: row = e + 8*(lane>>4), col = lane&15.
  const int lane = tid & 31;
  const int r = lane & 15, h = lane >> 4;
  #pragma unroll
  for (int mi = 0; mi < 4; ++mi) {
    #pragma unroll
    for (int ni = 0; ni < 2; ++ni) {
      const int col = tileN + wn * 32 + ni * 16 + r;
      const float bv = bias ? bias[col] : 0.0f;
      #pragma unroll
      for (int e = 0; e < 8; ++e) {
        const int row = tileM + wm * 64 + mi * 16 + e + 8 * h;
        float v = acc[mi][ni][e] + bv;
        if (FLAGS & GF_RELU) v = fmaxf(v, 0.0f);
        if (FLAGS & GF_VTR) {
          // V projection: write transposed per head, [b][h][hd][s]
          const int bb_ = row >> 10, ss = row & (Sc - 1);
          const int hh_ = col >> 6,  dd = col & (HDc - 1);
          ((bf16_t*)Cout)[(((size_t)bb_ * Hc + hh_) * HDc + dd) * Sc + ss] = (bf16_t)v;
        } else if (FLAGS & GF_OBF) {
          ((bf16_t*)Cout)[(size_t)row * N + col] = (bf16_t)v;
        } else {
          ((float*)Cout)[(size_t)row * N + col] = v;
        }
      }
    }
  }
}

// ---------------------------------------------------------------------------
// Flash attention with ALiBi + causal mask, bf16 inputs.
// Grid: (B*H, S/64). Block: 128 threads = 4 waves; wave w owns query rows
// [qt*64 + w*16, +16). HD=64 -> 2 WMMA K-chunks. K/V tiles arrive via async
// DMA; V comes pre-transposed ([b][h][hd][s]) so its B-fragments are
// contiguous rows.
// ---------------------------------------------------------------------------
#define ASTR 72   // LDS row stride (bf16); 144 B, fragments stay 16-B aligned

__global__ __launch_bounds__(128)
void attention_kernel(const bf16_t* __restrict__ Qb, const bf16_t* __restrict__ Kb,
                      const bf16_t* __restrict__ VbT, bf16_t* __restrict__ Ob)
{
  __shared__ __align__(16) bf16_t sQ [64 * ASTR];
  __shared__ __align__(16) bf16_t sK [64 * ASTR];
  __shared__ __align__(16) bf16_t sVT[64 * ASTR];      // [hd][key]
  __shared__ __align__(16) bf16_t sP [4 * 16 * ASTR];  // per-wave P tiles

  const int bh = blockIdx.x;
  const int b  = bh / Hc, hh = bh % Hc;
  const int qt = blockIdx.y;
  const int qbase = qt * 64;
  const int tid  = threadIdx.x;
  const int wave = tid >> 5;
  const int lane = tid & 31;
  const int r = lane & 15, hl = lane >> 4;

  const float scale = 0.125f;                          // 1/sqrt(64)
  const float slope = exp2f(-0.5f * (float)(hh + 1));  // 2^(-8*(h+1)/16)

  // Stage 64x64 Q tile via async DMA (512 chunks / 128 threads = 4 each).
  const bf16_t* Qp = Qb + ((size_t)b * Sc + qbase) * Dc + hh * HDc;
  #pragma unroll
  for (int i = 0; i < 4; ++i) {
    const int c = tid + i * 128;
    const int row = c >> 3, col8 = (c & 7) << 3;
    async_ld16(sQ + row * ASTR + col8, Qp + (size_t)row * Dc + col8);
  }
  asm volatile("s_wait_asynccnt 0x0" ::: "memory");
  __syncthreads();

  v16bf qa[2];
  #pragma unroll
  for (int kc = 0; kc < 2; ++kc)
    qa[kc] = load_frag_a(sQ + (wave * 16) * ASTR + kc * 32, ASTR);

  v8f oacc[4] = {};
  float mrow[8], lrow[8];
  #pragma unroll
  for (int e = 0; e < 8; ++e) { mrow[e] = -1e30f; lrow[e] = 0.0f; }

  for (int kt = 0; kt <= qt; ++kt) {
    __syncthreads();   // all waves done reading previous sK/sVT
    const bf16_t* Kp = Kb  + ((size_t)b * Sc + kt * 64) * Dc + hh * HDc;
    const bf16_t* Vp = VbT + ((size_t)b * Hc + hh) * (size_t)HDc * Sc + kt * 64;
    #pragma unroll
    for (int i = 0; i < 4; ++i) {
      const int c = tid + i * 128;
      const int row = c >> 3, col8 = (c & 7) << 3;
      async_ld16(sK  + row * ASTR + col8, Kp + (size_t)row * Dc + col8);
      async_ld16(sVT + row * ASTR + col8, Vp + (size_t)row * Sc + col8);
    }
    asm volatile("s_wait_asynccnt 0x0" ::: "memory");
    __syncthreads();

    // Scores: q(16x64) @ k^T(64keys) -> 4 N-frags.
    v8f sacc[4] = {};
    #pragma unroll
    for (int kc = 0; kc < 2; ++kc)
      #pragma unroll
      for (int nf = 0; nf < 4; ++nf) {
        v16bf kb16 = load_frag_b(sK + (nf * 16) * ASTR + kc * 32, ASTR);
        sacc[nf] = wmma_bf16(qa[kc], kb16, sacc[nf]);
      }

    // Online softmax with ALiBi + causal; row stats per half-wave.
    float pv[4][8];
    #pragma unroll
    for (int e = 0; e < 8; ++e) {
      const int qrow = qbase + wave * 16 + e + 8 * hl;
      float mx = -1e30f;
      #pragma unroll
      for (int nf = 0; nf < 4; ++nf) {
        const int kpos = kt * 64 + nf * 16 + r;
        float s = sacc[nf][e] * scale - slope * (float)(qrow - kpos);
        if (kpos > qrow) s = -1e30f;
        pv[nf][e] = s;
        mx = fmaxf(mx, s);
      }
      #pragma unroll
      for (int m = 1; m < 16; m <<= 1) mx = fmaxf(mx, __shfl_xor(mx, m, 32));
      const float mnew  = fmaxf(mrow[e], mx);
      const float alpha = __expf(mrow[e] - mnew);
      mrow[e] = mnew;
      float sum = 0.0f;
      #pragma unroll
      for (int nf = 0; nf < 4; ++nf) {
        const float p = __expf(pv[nf][e] - mnew);
        pv[nf][e] = p;
        sum += p;
      }
      #pragma unroll
      for (int m = 1; m < 16; m <<= 1) sum += __shfl_xor(sum, m, 32);
      lrow[e] = lrow[e] * alpha + sum;
      #pragma unroll
      for (int nf = 0; nf < 4; ++nf) oacc[nf][e] *= alpha;
    }

    // C-layout -> A-layout transpose of P through wave-private LDS.
    bf16_t* myP = sP + wave * 16 * ASTR;
    #pragma unroll
    for (int e = 0; e < 8; ++e) {
      const int prow = e + 8 * hl;
      #pragma unroll
      for (int nf = 0; nf < 4; ++nf)
        myP[prow * ASTR + nf * 16 + r] = (bf16_t)pv[nf][e];
    }
    asm volatile("s_wait_dscnt 0" ::: "memory");  // wave-local LDS WAR/RAW

    #pragma unroll
    for (int kc = 0; kc < 2; ++kc) {
      v16bf pa = load_frag_a(myP + kc * 32, ASTR);
      #pragma unroll
      for (int nf = 0; nf < 4; ++nf) {
        v16bf vb = load_frag_b(sVT + (nf * 16) * ASTR + kc * 32, ASTR);
        oacc[nf] = wmma_bf16(pa, vb, oacc[nf]);
      }
    }
  }

  bf16_t* Op = Ob + ((size_t)b * Sc + qbase) * Dc + hh * HDc;
  #pragma unroll
  for (int nf = 0; nf < 4; ++nf)
    #pragma unroll
    for (int e = 0; e < 8; ++e) {
      const int row = wave * 16 + e + 8 * hl;
      Op[(size_t)row * Dc + nf * 16 + r] = (bf16_t)(oacc[nf][e] / lrow[e]);
    }
}

// ---------------------------------------------------------------------------
// fp32 -> bf16 bulk convert (float4 in, 8-B packed out)
// ---------------------------------------------------------------------------
__global__ __launch_bounds__(256)
void cvt_bf16_kernel(const float* __restrict__ src, bf16_t* __restrict__ dst, int n4)
{
  const int i = blockIdx.x * 256 + threadIdx.x;
  if (i < n4) {
    float4 v = ((const float4*)src)[i];
    v4bf o = { (bf16_t)v.x, (bf16_t)v.y, (bf16_t)v.z, (bf16_t)v.w };
    *(v4bf*)(dst + (size_t)i * 4) = o;
  }
}

// ---------------------------------------------------------------------------
// Embedding gather * sqrt(D), dual fp32 + bf16 output
// ---------------------------------------------------------------------------
__global__ __launch_bounds__(256)
void embed_kernel(const int* __restrict__ ids, const float* __restrict__ emb,
                  float* __restrict__ x, bf16_t* __restrict__ xbf)
{
  const int row = blockIdx.x;
  const float* src = emb + (size_t)ids[row] * Dc;
  float* dst = x + (size_t)row * Dc;
  bf16_t* dbf = xbf + (size_t)row * Dc;
  for (int c = threadIdx.x; c < Dc; c += 256) {
    const float v = src[c] * 32.0f;    // sqrt(1024)
    dst[c] = v;
    dbf[c] = (bf16_t)v;
  }
}

// ---------------------------------------------------------------------------
// Fused residual-add + LayerNorm, dual fp32 + bf16 output.
// ---------------------------------------------------------------------------
__global__ __launch_bounds__(256)
void ln_kernel(const float* __restrict__ resid, const float* __restrict__ delta,
               const float* __restrict__ w, const float* __restrict__ bb,
               float* __restrict__ out, bf16_t* __restrict__ outbf)
{
  __shared__ float red0[8], red1[8];
  const int row = blockIdx.x;
  const float* rp = resid + (size_t)row * Dc;
  const float* dp = delta ? delta + (size_t)row * Dc : nullptr;

  float v[4], s = 0.0f, s2 = 0.0f;
  #pragma unroll
  for (int i = 0; i < 4; ++i) {
    const int c = threadIdx.x + i * 256;
    float t = rp[c] + (dp ? dp[c] : 0.0f);
    v[i] = t; s += t; s2 += t * t;
  }
  #pragma unroll
  for (int m = 1; m < 32; m <<= 1) { s += __shfl_xor(s, m, 32); s2 += __shfl_xor(s2, m, 32); }
  const int wave = threadIdx.x >> 5, lane = threadIdx.x & 31;
  if (lane == 0) { red0[wave] = s; red1[wave] = s2; }
  __syncthreads();
  if (wave == 0) {
    float a  = (lane < 8) ? red0[lane] : 0.0f;
    float a2 = (lane < 8) ? red1[lane] : 0.0f;
    #pragma unroll
    for (int m = 1; m < 8; m <<= 1) { a += __shfl_xor(a, m, 32); a2 += __shfl_xor(a2, m, 32); }
    if (lane == 0) { red0[0] = a; red1[0] = a2; }
  }
  __syncthreads();
  const float mean = red0[0] * (1.0f / Dc);
  const float var  = red1[0] * (1.0f / Dc) - mean * mean;
  const float rstd = rsqrtf(var + 1e-5f);
  #pragma unroll
  for (int i = 0; i < 4; ++i) {
    const int c = threadIdx.x + i * 256;
    const float o = (v[i] - mean) * rstd * w[c] + bb[c];
    out[(size_t)row * Dc + c] = o;
    outbf[(size_t)row * Dc + c] = (bf16_t)o;
  }
}

// ---------------------------------------------------------------------------
// Orchestration
// ---------------------------------------------------------------------------
extern "C" void kernel_launch(void* const* d_in, const int* in_sizes, int n_in,
                              void* d_out, int out_size, void* d_ws, size_t ws_size,
                              hipStream_t stream) {
  (void)in_sizes; (void)n_in; (void)out_size; (void)ws_size;

  const int*   ids   = (const int*)  d_in[0];
  const float* emb   = (const float*)d_in[1];
  const float* Wq    = (const float*)d_in[2];
  const float* bq    = (const float*)d_in[3];
  const float* Wk    = (const float*)d_in[4];
  const float* bk    = (const float*)d_in[5];
  const float* Wv    = (const float*)d_in[6];
  const float* bv    = (const float*)d_in[7];
  const float* Wo    = (const float*)d_in[8];
  const float* bo    = (const float*)d_in[9];
  const float* W1    = (const float*)d_in[10];
  const float* b1    = (const float*)d_in[11];
  const float* W2    = (const float*)d_in[12];
  const float* b2    = (const float*)d_in[13];
  const float* ln1w  = (const float*)d_in[14];
  const float* ln1b  = (const float*)d_in[15];
  const float* ln2w  = (const float*)d_in[16];
  const float* ln2b  = (const float*)d_in[17];
  const float* lnfw  = (const float*)d_in[18];
  const float* lnfb  = (const float*)d_in[19];
  const float* Whead = (const float*)d_in[20];

  const size_t AD = (size_t)MTOK * Dc;          // 2M activations
  char* p = (char*)d_ws;
  float*  x    = (float*) p; p += AD * 4;               // 8 MB
  float*  tb   = (float*) p; p += AD * 4;               // 8 MB
  bf16_t* xbf  = (bf16_t*)p; p += AD * 2;               // 4 MB
  bf16_t* qbf  = (bf16_t*)p; p += AD * 2;
  bf16_t* kbf  = (bf16_t*)p; p += AD * 2;
  bf16_t* vbT  = (bf16_t*)p; p += AD * 2;               // [b][h][hd][s]
  bf16_t* abf  = (bf16_t*)p; p += AD * 2;
  bf16_t* fbf  = (bf16_t*)p; p += (size_t)MTOK * DFFc * 2;      // 16 MB
  bf16_t* wsc  = (bf16_t*)p; p += (size_t)12 * 1024 * 1024 * 2; // per-layer weights
  bf16_t* whbf = (bf16_t*)p; p += (size_t)Vc * Dc * 2;  // 64 MB

  const size_t M1 = (size_t)Dc * Dc;            // 1M
  bf16_t* wqb = wsc;
  bf16_t* wkb = wsc + 1 * M1;
  bf16_t* wvb = wsc + 2 * M1;
  bf16_t* wob = wsc + 3 * M1;
  bf16_t* w1b = wsc + 4 * M1;                   // 4M
  bf16_t* w2b = wsc + 8 * M1;                   // 4M

  const dim3 blk(256);
  auto cvt = [&](const float* s, bf16_t* d, size_t n) {
    const int n4 = (int)(n / 4);
    cvt_bf16_kernel<<<(n4 + 255) / 256, blk, 0, stream>>>(s, d, n4);
  };

  embed_kernel<<<MTOK, blk, 0, stream>>>(ids, emb, x, xbf);
  cvt(Whead, whbf, (size_t)Vc * Dc);

  const dim3 gproj(Dc / GBN,   MTOK / GBM);     // (8, 16)
  const dim3 gff1 (DFFc / GBN, MTOK / GBM);     // (32, 16)
  const dim3 ghead(Vc / GBN,   MTOK / GBM);     // (250, 16)
  const dim3 gattn(Bc * Hc, Sc / 64);           // (32, 16)

  for (int l = 0; l < Lc; ++l) {
    const size_t wo = (size_t)l * M1;
    cvt(Wq + wo, wqb, M1);
    cvt(Wk + wo, wkb, M1);
    cvt(Wv + wo, wvb, M1);
    cvt(Wo + wo, wob, M1);
    cvt(W1 + (size_t)l * DFFc * Dc, w1b, (size_t)DFFc * Dc);
    cvt(W2 + (size_t)l * Dc * DFFc, w2b, (size_t)Dc * DFFc);

    gemm_bf16_kernel<GF_OBF><<<gproj, blk, 0, stream>>>(xbf, wqb, bq + (size_t)l * Dc, qbf, MTOK, Dc, Dc);
    gemm_bf16_kernel<GF_OBF><<<gproj, blk, 0, stream>>>(xbf, wkb, bk + (size_t)l * Dc, kbf, MTOK, Dc, Dc);
    gemm_bf16_kernel<GF_OBF | GF_VTR><<<gproj, blk, 0, stream>>>(xbf, wvb, bv + (size_t)l * Dc, vbT, MTOK, Dc, Dc);
    attention_kernel<<<gattn, dim3(128), 0, stream>>>(qbf, kbf, vbT, abf);
    gemm_bf16_kernel<0><<<gproj, blk, 0, stream>>>(abf, wob, bo + (size_t)l * Dc, tb, MTOK, Dc, Dc);
    ln_kernel<<<MTOK, blk, 0, stream>>>(x, tb, ln1w + (size_t)l * Dc, ln1b + (size_t)l * Dc, x, xbf);
    gemm_bf16_kernel<GF_OBF | GF_RELU><<<gff1, blk, 0, stream>>>(xbf, w1b, b1 + (size_t)l * DFFc, fbf, MTOK, DFFc, Dc);
    gemm_bf16_kernel<0><<<gproj, blk, 0, stream>>>(fbf, w2b, b2 + (size_t)l * Dc, tb, MTOK, Dc, DFFc);
    ln_kernel<<<MTOK, blk, 0, stream>>>(x, tb, ln2w + (size_t)l * Dc, ln2b + (size_t)l * Dc, x, xbf);
  }

  ln_kernel<<<MTOK, blk, 0, stream>>>(x, nullptr, lnfw, lnfb, x, xbf);
  gemm_bf16_kernel<0><<<ghead, blk, 0, stream>>>(xbf, whbf, nullptr, (float*)d_out, MTOK, Vc, Dc);
}
